// MultiheadAttention_4466765988168
// MI455X (gfx1250) — matrix operand order repaired and checked by reading, not compiled, not run on hardware
//
#include <hip/hip_runtime.h>
#include <math.h>

typedef __attribute__((ext_vector_type(16))) __bf16 v16bf;
typedef __attribute__((ext_vector_type(8)))  __bf16 v8bf;
typedef __attribute__((ext_vector_type(8)))  float  v8f;

#define DIM   512
#define NH    8
#define HD    64
#define BB    2
#define TT    4096
#define MROWS (BB*TT)     // 8192
#define NQKV  (3*DIM)     // 1536
#define SCALE 0.125f      // HEAD_DIM^-0.5

// ---------------------------------------------------------------------------
// WMMA helpers (contiguous k order: element i of lane half h is k = 16*h + i, wave32)
// ---------------------------------------------------------------------------
__device__ __forceinline__ v8f wmma_bf16(v16bf a, v16bf b, v8f c) {
  // (neg_a, A, neg_b, B, c_mod, C, reuse_a, reuse_b)
  return __builtin_amdgcn_wmma_f32_16x16x32_bf16(false, a, false, b,
                                                 (short)0, c, false, false);
}

// A operand: 16x32 row-major tile at (mbase, kbase), row stride ldk elements.
// lane m (0..15): e0..15 = A[m][kb+half*16+i]
__device__ __forceinline__ v16bf load_a_frag(const __bf16* __restrict__ base,
                                             int ldk, int mbase, int kbase,
                                             int lane) {
  int half = lane >> 4;
  int row  = mbase + (lane & 15);
  const __bf16* p = base + (size_t)row * ldk + kbase + half * 16;
  v8bf lo = *(const v8bf*)p;
  v8bf hi = *(const v8bf*)(p + 8);
  v16bf a;
#pragma unroll
  for (int i = 0; i < 8; ++i) { a[i] = lo[i]; a[i + 8] = hi[i]; }
  return a;
}

// B operand from B^T stored row-major [N][K]: lane n holds column (nbase+n),
// e_i = B[kbase + half*16 + i][n]  ==  BT[nbase+n][kbase+half*16+i]
// -> one contiguous 32-byte read per lane.
__device__ __forceinline__ v16bf load_b_frag(const __bf16* __restrict__ bt,
                                             int ldk, int nbase, int kbase,
                                             int lane) {
  int half = lane >> 4;
  int col  = nbase + (lane & 15);
  const __bf16* p = bt + (size_t)col * ldk + kbase + half * 16;
  return *(const v16bf*)p;
}

// ---------------------------------------------------------------------------
// Precision conversion kernels
// ---------------------------------------------------------------------------
__global__ void cvt_bf16_kernel(const float* __restrict__ src,
                                __bf16* __restrict__ dst, int n) {
  int i = blockIdx.x * 256 + threadIdx.x;
  if (i < n) dst[i] = (__bf16)src[i];
}

__global__ void transpose_cvt_kernel(const float* __restrict__ src,
                                     __bf16* __restrict__ dst,
                                     int rows, int cols) {
  int i = blockIdx.x * 256 + threadIdx.x;
  if (i < rows * cols) {
    int r = i / cols, c = i % cols;
    dst[(size_t)c * rows + r] = (__bf16)src[i];
  }
}

// ---------------------------------------------------------------------------
// QKV projection: qkv = xb @ w_qkv ; scatter into Q[B,H,T,64], K[B,H,T,64],
// VT[B,H,64,T] (V transposed so PV B-operand is contiguous).
// Block = 4 waves; wave = 16(M)x64(N) tile; grid (M/64, 1536/64).
// ---------------------------------------------------------------------------
__global__ __launch_bounds__(128)
void qkv_gemm_kernel(const __bf16* __restrict__ xb,     // [8192][512]
                     const __bf16* __restrict__ wqkvT,  // [1536][512]
                     __bf16* __restrict__ Q,
                     __bf16* __restrict__ Km,
                     __bf16* __restrict__ VT) {
  int lane  = threadIdx.x & 31, wave = threadIdx.x >> 5;
  int mbase = blockIdx.x * 64 + wave * 16;
  int nbase = blockIdx.y * 64;
  v8f acc[4] = {};
  for (int kb = 0; kb < DIM; kb += 32) {
    v16bf a = load_a_frag(xb, DIM, mbase, kb, lane);
#pragma unroll
    for (int j = 0; j < 4; ++j) {
      v16bf b = load_b_frag(wqkvT, DIM, nbase + j * 16, kb, lane);
      acc[j] = wmma_bf16(a, b, acc[j]);
    }
  }
  int half = lane >> 4, c = lane & 15;
#pragma unroll
  for (int j = 0; j < 4; ++j) {
    int gn = nbase + j * 16 + c;
    int which = gn >> 9;           // 0=Q, 1=K, 2=V
    int h  = (gn >> 6) & 7;
    int hd = gn & 63;
#pragma unroll
    for (int r = 0; r < 8; ++r) {
      int mrow = mbase + r + 8 * half;
      int b = mrow >> 12, t = mrow & (TT - 1);
      int bh = b * NH + h;
      __bf16 val = (__bf16)acc[j][r];
      if (which == 0)      Q [((size_t)bh * TT + t) * HD + hd] = val;
      else if (which == 1) Km[((size_t)bh * TT + t) * HD + hd] = val;
      else                 VT[((size_t)bh * HD + hd) * TT + t] = val;
    }
  }
}

// ---------------------------------------------------------------------------
// Flash attention: one block (4 waves) per (b,h, 64-row q-tile); each wave
// owns 16 q rows. Online softmax in f32; S and PV via bf16 WMMA.
// ---------------------------------------------------------------------------
__global__ __launch_bounds__(128)
void attn_kernel(const __bf16* __restrict__ Q,
                 const __bf16* __restrict__ Km,
                 const __bf16* __restrict__ VT,
                 const int* __restrict__ mask,
                 __bf16* __restrict__ attnb) {            // [8192][512]
  __shared__ __align__(64) __bf16 plds[4][16][32];        // wave-private P tile
  const float INF = __builtin_inff();

  int lane = threadIdx.x & 31, wave = threadIdx.x >> 5;
  int half = lane >> 4, c = lane & 15;
  int bh = blockIdx.x >> 6;                 // 64 q-tiles of 64 rows per (b,h)
  int qt = (blockIdx.x & 63) * 64 + wave * 16;

  const __bf16* qp = Q  + (size_t)bh * TT * HD;
  const __bf16* kp = Km + (size_t)bh * TT * HD;
  const __bf16* vp = VT + (size_t)bh * HD * TT;

  v16bf aq0 = load_a_frag(qp, HD, qt, 0,  lane);          // d = 0..31
  v16bf aq1 = load_a_frag(qp, HD, qt, 32, lane);          // d = 32..63

  float mrow[8], lrow[8];
  int   rowg[8];
  v8f   acc[4] = {};
#pragma unroll
  for (int r = 0; r < 8; ++r) {
    mrow[r] = -INF; lrow[r] = 0.f;
    rowg[r] = qt + r + 8 * half;
  }

  for (int kb = 0; kb < TT; kb += 32) {
    __builtin_prefetch(kp + (size_t)(kb + 32) * HD, 0, 3);
    __builtin_prefetch(kp + (size_t)(kb + 48) * HD, 0, 3);

    // ---- S = Q @ K^T for 32 keys (two 16-key subtiles) ----
    v8f s0 = {}, s1 = {};
    {
      v16bf b00 = load_b_frag(kp, HD, kb,      0,  lane);
      v16bf b01 = load_b_frag(kp, HD, kb,      32, lane);
      v16bf b10 = load_b_frag(kp, HD, kb + 16, 0,  lane);
      v16bf b11 = load_b_frag(kp, HD, kb + 16, 32, lane);
      s0 = wmma_bf16(aq0, b00, s0);  s0 = wmma_bf16(aq1, b01, s0);
      s1 = wmma_bf16(aq0, b10, s1);  s1 = wmma_bf16(aq1, b11, s1);
    }

    // ---- scale + mask (C-layout: lane holds rows r+8*half at col c) ----
    float v0[8], v1[8], al[8];
#pragma unroll
    for (int r = 0; r < 8; ++r) {
      int m0 = mask[(size_t)rowg[r] * TT + kb + c];
      int m1 = mask[(size_t)rowg[r] * TT + kb + 16 + c];
      v0[r] = m0 ? s0[r] * SCALE : -INF;
      v1[r] = m1 ? s1[r] * SCALE : -INF;
    }

    // ---- online softmax: row reductions over 16 lanes of this half ----
#pragma unroll
    for (int r = 0; r < 8; ++r) {
      float tmax = fmaxf(v0[r], v1[r]);
#pragma unroll
      for (int off = 1; off < 16; off <<= 1)
        tmax = fmaxf(tmax, __shfl_xor(tmax, off, 32));
      float mnew  = fmaxf(mrow[r], tmax);
      float alpha = (mrow[r] == -INF) ? 0.f : __expf(mrow[r] - mnew);
      float p0 = (v0[r] == -INF) ? 0.f : __expf(v0[r] - mnew);
      float p1 = (v1[r] == -INF) ? 0.f : __expf(v1[r] - mnew);
      float rsum = p0 + p1;
#pragma unroll
      for (int off = 1; off < 16; off <<= 1)
        rsum += __shfl_xor(rsum, off, 32);
      lrow[r] = lrow[r] * alpha + rsum;
      mrow[r] = mnew;
      al[r] = alpha;
      v0[r] = p0; v1[r] = p1;
    }

    // ---- rescale accumulators ----
#pragma unroll
    for (int j = 0; j < 4; ++j)
#pragma unroll
      for (int r = 0; r < 8; ++r)
        acc[j][r] *= al[r];

    // ---- P: C-layout -> LDS row-major -> A-operand layout ----
    __syncthreads();
#pragma unroll
    for (int r = 0; r < 8; ++r) {
      plds[wave][r + 8 * half][c]      = (__bf16)v0[r];
      plds[wave][r + 8 * half][c + 16] = (__bf16)v1[r];
    }
    __syncthreads();
    v16bf ap = load_a_frag(&plds[wave][0][0], 32, 0, 0, lane);

    // ---- O += P @ V  (VT rows are d, contiguous over keys) ----
#pragma unroll
    for (int j = 0; j < 4; ++j) {
      v16bf bv = load_b_frag(vp, TT, j * 16, kb, lane);
      acc[j] = wmma_bf16(ap, bv, acc[j]);
    }
  }

  // ---- normalize + store attn in [b, t, h*64+d] (bf16) ----
  float inv[8];
#pragma unroll
  for (int r = 0; r < 8; ++r) inv[r] = lrow[r] > 0.f ? 1.f / lrow[r] : 0.f;
  int h = bh & 7, b = bh >> 3;
#pragma unroll
  for (int j = 0; j < 4; ++j)
#pragma unroll
    for (int r = 0; r < 8; ++r) {
      int t = qt + r + 8 * half;
      attnb[((size_t)(b * TT + t)) * DIM + h * HD + j * 16 + c] =
          (__bf16)(acc[j][r] * inv[r]);
    }
}

// ---------------------------------------------------------------------------
// Output projection: out = attnb @ w_out^T + b_out (w_out row-major IS B^T).
// ---------------------------------------------------------------------------
__global__ __launch_bounds__(128)
void out_proj_kernel(const __bf16* __restrict__ attnb,  // [8192][512]
                     const __bf16* __restrict__ wob,    // [512][512]
                     const float* __restrict__ bout,
                     float* __restrict__ out) {
  int lane  = threadIdx.x & 31, wave = threadIdx.x >> 5;
  int mbase = blockIdx.x * 64 + wave * 16;
  int nbase = blockIdx.y * 64;
  v8f acc[4] = {};
  for (int kb = 0; kb < DIM; kb += 32) {
    v16bf a = load_a_frag(attnb, DIM, mbase, kb, lane);
#pragma unroll
    for (int j = 0; j < 4; ++j) {
      v16bf b = load_b_frag(wob, DIM, nbase + j * 16, kb, lane);
      acc[j] = wmma_bf16(a, b, acc[j]);
    }
  }
  int half = lane >> 4, c = lane & 15;
#pragma unroll
  for (int j = 0; j < 4; ++j) {
    int gn = nbase + j * 16 + c;
    float bv = bout[gn];
#pragma unroll
    for (int r = 0; r < 8; ++r) {
      int mrow = mbase + r + 8 * half;
      out[(size_t)mrow * DIM + gn] = acc[j][r] + bv;
    }
  }
}

// ---------------------------------------------------------------------------
extern "C" void kernel_launch(void* const* d_in, const int* in_sizes, int n_in,
                              void* d_out, int out_size, void* d_ws,
                              size_t ws_size, hipStream_t stream) {
  const float* x     = (const float*)d_in[0];   // [2,4096,512]
  const int*   mask  = (const int*)d_in[1];     // [4096,4096]
  const float* w_qkv = (const float*)d_in[2];   // [512,1536]
  const float* w_out = (const float*)d_in[3];   // [512,512]
  const float* b_out = (const float*)d_in[4];   // [512]
  float* out = (float*)d_out;

  char* ws = (char*)d_ws;
  const size_t MB = 1024 * 1024;
  __bf16* xb    = (__bf16*)(ws);                 // 8 MB  [8192][512]
  __bf16* wqkvT = (__bf16*)(ws +  8 * MB);       // 1.5MB [1536][512]
  __bf16* wob   = (__bf16*)(ws + 10 * MB);       // 0.5MB [512][512]
  __bf16* Q     = (__bf16*)(ws + 11 * MB);       // 8 MB  [2,8,4096,64]
  __bf16* Km    = (__bf16*)(ws + 19 * MB);       // 8 MB
  __bf16* VT    = (__bf16*)(ws + 27 * MB);       // 8 MB  [2,8,64,4096]
  __bf16* attnb = (__bf16*)(ws + 35 * MB);       // 8 MB  [8192][512]

  int nx = MROWS * DIM;                          // 4194304
  cvt_bf16_kernel<<<(nx + 255) / 256, 256, 0, stream>>>(x, xb, nx);
  transpose_cvt_kernel<<<(DIM * NQKV + 255) / 256, 256, 0, stream>>>(
      w_qkv, wqkvT, DIM, NQKV);
  cvt_bf16_kernel<<<(DIM * DIM + 255) / 256, 256, 0, stream>>>(
      w_out, wob, DIM * DIM);

  qkv_gemm_kernel<<<dim3(MROWS / 64, NQKV / 64), 128, 0, stream>>>(
      xb, wqkvT, Q, Km, VT);

  attn_kernel<<<BB * NH * (TT / 64), 128, 0, stream>>>(Q, Km, VT, mask, attnb);

  out_proj_kernel<<<dim3(MROWS / 64, DIM / 64), 128, 0, stream>>>(
      attnb, wob, b_out, out);
}
